// HUNYUAN_45981919871402
// MI455X (gfx1250) — matrix operand order repaired and checked by reading, not compiled, not run on hardware
//
#include <hip/hip_runtime.h>
#include <cstdint>

// ---------------- types for CDNA5 WMMA / TDM ----------------
typedef __attribute__((ext_vector_type(16))) __bf16 v16bf;
typedef __attribute__((ext_vector_type(8)))  float  v8f;
typedef __attribute__((ext_vector_type(4)))  unsigned int u32x4;
typedef __attribute__((ext_vector_type(8)))  int i32x8;
typedef __attribute__((ext_vector_type(4)))  int i32x4;

// ---------------- model constants ----------------
#define L_    4
#define NH_   16
#define KVH_  4
#define HD_   128
#define H_    2048
#define F_    8192
#define V_    32000
#define G_    4
#define T_    256
#define HIST_ 1024
#define KV_   1280
#define EPS_  1e-6f
#define QKSCALE_ 0.29730177875068026f   // 128^-0.25

// ---------------- TDM 2D tile load: global -> LDS (async, TENSORcnt) ----------------
// D# packing per CDNA5 ISA. data_size=4B, 2D tile,
// pad_enable adds LDS padding after each tile row (pad_amount 1 => +2 dwords)
// so LDS row stride = tile_w + 2 floats.
// 6-arg builtin form (clang-23 / therock): (g0, g1, g2, g3, g_extra, cpol)
__device__ __forceinline__ void tdm_load_2d(unsigned lds_addr, const float* gptr,
                                            int tile_w, int tile_h, long row_stride,
                                            unsigned pad_interval) {
    unsigned long long ga = (unsigned long long)(uintptr_t)gptr;
    u32x4 g0;
    g0[0] = 1u;                                   // count=1, user mode, no gather
    g0[1] = lds_addr;                             // lds_addr [63:32]
    g0[2] = (unsigned)ga;                         // global_addr low
    g0[3] = ((unsigned)(ga >> 32) & 0x01FFFFFFu) | (2u << 30);  // addr hi | type=2
    i32x8 g1;
    g1[0] = (int)((2u << 16) | (1u << 20) | (pad_interval << 22) | (1u << 25)); // data_size=4B, pad_en, pad_amount=2dw
    g1[1] = (int)(((unsigned)tile_w & 0xFFFFu) << 16);            // tensor_dim0 lo (== tile_w)
    g1[2] = (int)((((unsigned)tile_w) >> 16) | (((unsigned)tile_h & 0xFFFFu) << 16)); // dim0 hi | tensor_dim1 lo
    g1[3] = (int)((((unsigned)tile_h) >> 16) | (((unsigned)tile_w) << 16));           // dim1 hi | tile_dim0
    g1[4] = (int)((unsigned)tile_h & 0xFFFFu);                    // tile_dim1 (tile_dim2=0)
    unsigned long long st = (unsigned long long)row_stride;
    g1[5] = (int)(unsigned)st;                                    // tensor_dim0_stride lo
    g1[6] = (int)((unsigned)(st >> 32) & 0xFFFFu);                // stride hi (dim1_stride=0)
    g1[7] = 0;
    i32x4 gz = {0, 0, 0, 0};
    i32x8 gz8 = {0, 0, 0, 0, 0, 0, 0, 0};
    __builtin_amdgcn_tensor_load_to_lds(g0, g1, gz, gz, gz8, 0);
}

// ---------------- 16-lane row reductions (rows live across lanes 0..15 / 16..31) ----------------
__device__ __forceinline__ float rowmax16(float v) {
    for (int m = 8; m >= 1; m >>= 1) v = fmaxf(v, __shfl_xor(v, m));
    return v;
}
__device__ __forceinline__ float rowsum16(float v) {
    for (int m = 8; m >= 1; m >>= 1) v += __shfl_xor(v, m);
    return v;
}

// =====================================================================
// Embedding dequant:  h[t][c] = (float)q * scale[id] + zp[id]
// =====================================================================
__global__ void hy_embed_kernel(const int* __restrict__ ids, const int* __restrict__ eq,
                                const float* __restrict__ esc, const float* __restrict__ ezp,
                                float* __restrict__ h) {
    int t = blockIdx.x;
    int id = ids[t];
    float sc = esc[id], zp = ezp[id];
    for (int c = threadIdx.x; c < H_; c += blockDim.x)
        h[(long)t * H_ + c] = (float)eq[(long)id * H_ + c] * sc + zp;
}

// =====================================================================
// RMSNorm over `width` per row:  y = x * rsqrt(mean(x^2)+eps) * w
// =====================================================================
__global__ void hy_rms_kernel(const float* __restrict__ x, const float* __restrict__ w,
                              float* __restrict__ y, int width) {
    int r = blockIdx.x;
    __shared__ float red[256];
    float s = 0.f;
    for (int c = threadIdx.x; c < width; c += 256) {
        float v = x[(long)r * width + c];
        s += v * v;
    }
    red[threadIdx.x] = s;
    __syncthreads();
    for (int st = 128; st > 0; st >>= 1) {
        if (threadIdx.x < st) red[threadIdx.x] += red[threadIdx.x + st];
        __syncthreads();
    }
    float inv = rsqrtf(red[0] / (float)width + EPS_);
    for (int c = threadIdx.x; c < width; c += 256)
        y[(long)r * width + c] = x[(long)r * width + c] * inv * w[c];
}

// =====================================================================
// GEMM: C[M,N] (+)= A[M,K] * B[K,N]; fp32 in HBM, bf16 WMMA, fp32 accum.
// Block tile 128x64, 8 waves (2x4), K-step 32, TDM double-buffered staging.
// Fragment loads are batched into local arrays first so the compiler can
// issue the whole DS-load group before a single s_wait_dscnt.
// =====================================================================
#define TILE_M 128
#define TILE_N 64
#define KSTEP  32
#define A_LD   34            // 32 + 2 pad dwords (TDM pad)
#define B_LD   66            // 64 + 2 pad dwords
#define A_SZ   (TILE_M * A_LD)   // 4352 floats
#define B_SZ   (KSTEP * B_LD)    // 2112 floats
#define BUF_SZ (A_SZ + B_SZ)     // 6464 floats

__global__ __launch_bounds__(256) void hy_gemm_kernel(const float* __restrict__ A,
                                                      const float* __restrict__ B,
                                                      float* __restrict__ C,
                                                      int N, int K, int accumulate) {
    __shared__ float lds[2 * BUF_SZ];
    const int tid = threadIdx.x;
    const int wid = tid >> 5;
    const int lane = tid & 31;
    const int h = lane >> 4, ln = lane & 15;
    const int wm = wid >> 2, wn = wid & 3;
    const int m0 = blockIdx.y * TILE_M;
    const int n0 = blockIdx.x * TILE_N;
    const int ksteps = K / KSTEP;

    v8f acc[4];
    v8f zero = {0.f, 0.f, 0.f, 0.f, 0.f, 0.f, 0.f, 0.f};
#pragma unroll
    for (int t = 0; t < 4; t++) acc[t] = zero;

    auto issue = [&](int buf, int ks) {
        unsigned ldsA = (unsigned)(uintptr_t)(&lds[buf * BUF_SZ]);
        unsigned ldsB = (unsigned)(uintptr_t)(&lds[buf * BUF_SZ + A_SZ]);
        // A tile: 128 rows x 32 floats, row stride K; pad every 32 dwords (code 4)
        tdm_load_2d(ldsA, A + (long)m0 * K + (long)ks * KSTEP, KSTEP, TILE_M, K, 4u);
        // B tile: 32 rows x 64 floats, row stride N; pad every 64 dwords (code 5)
        tdm_load_2d(ldsB, B + (long)ks * KSTEP * N + n0, TILE_N, KSTEP, N, 5u);
    };

    if (wid == 0) issue(0, 0);
    int buf = 0;
    for (int ks = 0; ks < ksteps; ++ks) {
        __syncthreads();                    // previous compute done before buf^1 is overwritten
        if (wid == 0) {
            if (ks + 1 < ksteps) {
                issue(buf ^ 1, ks + 1);
                __builtin_amdgcn_s_wait_tensorcnt(2);   // oldest pair (this step) complete
            } else {
                __builtin_amdgcn_s_wait_tensorcnt(0);
            }
        }
        __syncthreads();                    // buf is ready for all waves
        const float* At = &lds[buf * BUF_SZ];
        const float* Bt = At + A_SZ;

        // ---- B fragment: batch 16 LDS loads, then convert ----
        float tb[16];
#pragma unroll
        for (int e = 0; e < 16; e++)
            tb[e] = Bt[((e < 8) ? (8 * h + e) : (16 + 8 * h + (e - 8))) * B_LD + wn * 16 + ln];
        v16bf bfr;
#pragma unroll
        for (int e = 0; e < 16; e++) bfr[e] = (__bf16)tb[e];

#pragma unroll
        for (int t = 0; t < 4; t++) {
            const float* arow = &At[(wm * 64 + t * 16 + ln) * A_LD];
            // ---- A fragment: batch 8 paired LDS loads, then convert ----
            float2 ta[8];
#pragma unroll
            for (int g = 0; g < 8; g++) {
                int k0 = (g < 4) ? (8 * h + 2 * g) : (16 + 8 * h + 2 * (g - 4));
                ta[g] = *(const float2*)&arow[k0];
            }
            v16bf afr;
#pragma unroll
            for (int g = 0; g < 8; g++) {
                afr[2 * g] = (__bf16)ta[g].x;
                afr[2 * g + 1] = (__bf16)ta[g].y;
            }
            acc[t] = __builtin_amdgcn_wmma_f32_16x16x32_bf16(false, afr, false, bfr,
                                                             (short)0, acc[t], false, false);
        }
        buf ^= 1;
    }
#pragma unroll
    for (int t = 0; t < 4; t++) {
#pragma unroll
        for (int i = 0; i < 8; i++) {
            int m = m0 + wm * 64 + t * 16 + i + 8 * h;
            int n = n0 + wn * 16 + ln;
            long off = (long)m * N + n;
            float v = acc[t][i];
            if (accumulate) C[off] += v; else C[off] = v;
        }
    }
}

// =====================================================================
// Rope + per-head RMS for Q (in place).  grid = T*NH, block = 128
// =====================================================================
__global__ void hy_rope_q_kernel(float* __restrict__ q, const float* __restrict__ qlnw) {
    int t = blockIdx.x / NH_, hh = blockIdx.x % NH_;
    int d = threadIdx.x;
    __shared__ float xs[HD_];
    __shared__ float red[HD_];
    float x = q[(long)t * H_ + hh * HD_ + d];
    xs[d] = x;
    __syncthreads();
    float rh = (d < 64) ? -xs[d + 64] : xs[d - 64];
    int j = d & 63;
    float freq = __powf(10000.f, -(float)j / 64.f);
    float ang = (float)(HIST_ + t) * freq;
    float c = (float)(_Float16)cosf(ang);   // fp16 round-trip like the reference table
    float sn = (float)(_Float16)sinf(ang);
    float y = x * c + rh * sn;
    red[d] = y * y;
    __syncthreads();
    for (int st = 64; st > 0; st >>= 1) {
        if (d < st) red[d] += red[d + st];
        __syncthreads();
    }
    float inv = rsqrtf(red[0] / (float)HD_ + EPS_);
    q[(long)t * H_ + hh * HD_ + d] = y * inv * (qlnw[d] * QKSCALE_);
}

// =====================================================================
// Rope + per-head RMS for K, written transposed into nk[kvh][hd][HIST+t]
// grid = T*KVH, block = 128
// =====================================================================
__global__ void hy_rope_k_kernel(const float* __restrict__ kbuf, const float* __restrict__ klnw,
                                 float* __restrict__ nkL) {
    int t = blockIdx.x / KVH_, kvh = blockIdx.x % KVH_;
    int d = threadIdx.x;
    __shared__ float xs[HD_];
    __shared__ float red[HD_];
    float x = kbuf[(long)t * (KVH_ * HD_) + kvh * HD_ + d];
    xs[d] = x;
    __syncthreads();
    float rh = (d < 64) ? -xs[d + 64] : xs[d - 64];
    int j = d & 63;
    float freq = __powf(10000.f, -(float)j / 64.f);
    float ang = (float)(HIST_ + t) * freq;
    float c = (float)(_Float16)cosf(ang);
    float sn = (float)(_Float16)sinf(ang);
    float y = x * c + rh * sn;
    red[d] = y * y;
    __syncthreads();
    for (int st = 64; st > 0; st >>= 1) {
        if (d < st) red[d] += red[d + st];
        __syncthreads();
    }
    float inv = rsqrtf(red[0] / (float)HD_ + EPS_);
    nkL[(long)kvh * HD_ * KV_ + (long)d * KV_ + HIST_ + t] = y * inv * (klnw[d] * QKSCALE_);
}

// =====================================================================
// Cache concat copies
// =====================================================================
__global__ void hy_copy_kcache_kernel(const float* __restrict__ kc, float* __restrict__ nkL) {
    long i = (long)blockIdx.x * 256 + threadIdx.x;            // KVH*HD*HIST elems
    long kvh = i / ((long)HD_ * HIST_);
    long rem = i % ((long)HD_ * HIST_);
    long hd = rem / HIST_, pos = rem % HIST_;
    nkL[kvh * HD_ * KV_ + hd * KV_ + pos] = kc[i];
}
__global__ void hy_copy_vcache_kernel(const float* __restrict__ vc, float* __restrict__ nvL) {
    long i = (long)blockIdx.x * 256 + threadIdx.x;            // KVH*HIST*HD elems
    long kvh = i / ((long)HIST_ * HD_);
    long rem = i % ((long)HIST_ * HD_);
    nvL[kvh * (long)KV_ * HD_ + rem] = vc[i];                 // rows [0,HIST) contiguous per kvh
}
__global__ void hy_write_vnew_kernel(const float* __restrict__ vbuf, float* __restrict__ nvL) {
    long i = (long)blockIdx.x * 256 + threadIdx.x;            // T*KVH*HD elems
    long t = i / (KVH_ * HD_);
    long rem = i % (KVH_ * HD_);
    long kvh = rem / HD_, hd = rem % HD_;
    nvL[kvh * (long)KV_ * HD_ + (HIST_ + t) * (long)HD_ + hd] = vbuf[i];
}

// =====================================================================
// Flash-style attention, 1 wave = (head, 16-row tile), WMMA QK^T and PV,
// online softmax with the reference's additive -128*flag mask.
// grid = NH*T/16/4 blocks, 128 threads (4 waves).
// =====================================================================
__global__ __launch_bounds__(128) void hy_attn_kernel(const float* __restrict__ q,
                                                      const float* __restrict__ nkL,
                                                      const float* __restrict__ nvL,
                                                      const float* __restrict__ flagp,
                                                      float* __restrict__ out) {
    const int wv = threadIdx.x >> 5;
    const int gw = blockIdx.x * 4 + wv;
    const int hh = gw >> 4;           // head
    const int rt = gw & 15;           // 16-row tile of T
    const int kvh = hh / G_;
    const int lane = threadIdx.x & 31, h = lane >> 4, ln = lane & 15;
    const float flag = flagp[0];
    const float* kp = nkL + (long)kvh * HD_ * KV_;   // [HD][KV]
    const float* vp = nvL + (long)kvh * KV_ * HD_;   // [KV][HD]

    __shared__ __attribute__((aligned(8))) __bf16 ptile[4][16 * 32];
    __bf16* pt = ptile[wv];

    // preload Q A-fragments (4 hd-chunks of 32); batch loads then convert
    v16bf aq[4];
#pragma unroll
    for (int c = 0; c < 4; c++) {
        const float* qrow = &q[(long)(rt * 16 + ln) * H_ + hh * HD_ + c * 32];
        float2 tq[8];
#pragma unroll
        for (int g = 0; g < 8; g++) {
            int k0 = (g < 4) ? (8 * h + 2 * g) : (16 + 8 * h + 2 * (g - 4));
            tq[g] = *(const float2*)&qrow[k0];
        }
#pragma unroll
        for (int g = 0; g < 8; g++) {
            aq[c][2 * g] = (__bf16)tq[g].x;
            aq[c][2 * g + 1] = (__bf16)tq[g].y;
        }
    }

    v8f o[8];
    v8f zero = {0.f, 0.f, 0.f, 0.f, 0.f, 0.f, 0.f, 0.f};
#pragma unroll
    for (int f = 0; f < 8; f++) o[f] = zero;
    float mrun[8], lrun[8];
#pragma unroll
    for (int v = 0; v < 8; v++) { mrun[v] = -1e30f; lrun[v] = 0.f; }

    for (int kc = 0; kc < KV_ / 32; ++kc) {
        int kv0 = kc * 32;
        float ps[2][8];
#pragma unroll
        for (int jj = 0; jj < 2; jj++) {
            v8f s = zero;
#pragma unroll
            for (int c = 0; c < 4; c++) {
                // batch 16 global loads of K^T, then convert
                float tk[16];
#pragma unroll
                for (int e = 0; e < 16; e++)
                    tk[e] = kp[(long)(c * 32 + ((e < 8) ? (8 * h + e) : (16 + 8 * h + (e - 8)))) * KV_ + kv0 + jj * 16 + ln];
                v16bf bk;
#pragma unroll
                for (int e = 0; e < 16; e++) bk[e] = (__bf16)tk[e];
                s = __builtin_amdgcn_wmma_f32_16x16x32_bf16(false, aq[c], false, bk,
                                                            (short)0, s, false, false);
            }
#pragma unroll
            for (int v = 0; v < 8; v++) {
                int r = rt * 16 + v + 8 * h;
                int col = kv0 + jj * 16 + ln;
                ps[jj][v] = s[v] + ((col <= r) ? 0.f : -128.f * flag);
            }
        }
        // online softmax update per row
#pragma unroll
        for (int v = 0; v < 8; v++) {
            float chm = rowmax16(fmaxf(ps[0][v], ps[1][v]));
            float mnew = fmaxf(mrun[v], chm);
            float alpha = __expf(mrun[v] - mnew);
            float p0 = __expf(ps[0][v] - mnew);
            float p1 = __expf(ps[1][v] - mnew);
            lrun[v] = lrun[v] * alpha + rowsum16(p0 + p1);
            mrun[v] = mnew;
#pragma unroll
            for (int f = 0; f < 8; f++) o[f][v] *= alpha;
            pt[(v + 8 * h) * 32 + ln] = (__bf16)p0;
            pt[(v + 8 * h) * 32 + 16 + ln] = (__bf16)p1;
        }
        __syncthreads();
        // P tile (16x32) back as A-fragment: batch the 8 dword LDS reads
        unsigned tp[8];
#pragma unroll
        for (int g = 0; g < 8; g++) {
            int k0 = (g < 4) ? (8 * h + 2 * g) : (16 + 8 * h + 2 * (g - 4));
            tp[g] = *(const unsigned*)&pt[ln * 32 + k0];
        }
        v16bf pa;
#pragma unroll
        for (int g = 0; g < 8; g++) {
            union { unsigned u; __bf16 b[2]; } cv;
            cv.u = tp[g];
            pa[2 * g] = cv.b[0];
            pa[2 * g + 1] = cv.b[1];
        }
        // O += P @ V  (batch V loads per fragment, then convert)
#pragma unroll
        for (int f = 0; f < 8; f++) {
            float tv[16];
#pragma unroll
            for (int e = 0; e < 16; e++)
                tv[e] = vp[(long)(kv0 + ((e < 8) ? (8 * h + e) : (16 + 8 * h + (e - 8)))) * HD_ + f * 16 + ln];
            v16bf bv;
#pragma unroll
            for (int e = 0; e < 16; e++) bv[e] = (__bf16)tv[e];
            o[f] = __builtin_amdgcn_wmma_f32_16x16x32_bf16(false, pa, false, bv,
                                                           (short)0, o[f], false, false);
        }
        __syncthreads();
    }
    // epilogue: divide by softmax denominator, write [T, NH*HD] head-major
#pragma unroll
    for (int v = 0; v < 8; v++) {
        float inv = 1.f / lrun[v];
        int r = rt * 16 + v + 8 * h;
#pragma unroll
        for (int f = 0; f < 8; f++)
            out[(long)r * H_ + hh * HD_ + f * 16 + ln] = o[f][v] * inv;
    }
}

// =====================================================================
// SwiGLU elementwise:  g = silu(g) * u
// =====================================================================
__global__ void hy_silu_mul_kernel(float* __restrict__ g, const float* __restrict__ u, long n) {
    long i = (long)blockIdx.x * 256 + threadIdx.x;
    if (i < n) {
        float x = g[i];
        g[i] = (x / (1.f + __expf(-x))) * u[i];
    }
}

// =====================================================================
// lm_head GEMV + block-level argmax partials, then final argmax.
// =====================================================================
__global__ void hy_logits_kernel(const float* __restrict__ hl, const float* __restrict__ w,
                                 float* __restrict__ bval, int* __restrict__ bidx) {
    int c = blockIdx.x * 256 + threadIdx.x;
    float acc = 0.f;
    for (int k = 0; k < H_; k++) acc = fmaf(hl[k], w[(long)k * V_ + c], acc);
    __shared__ float sv[256];
    __shared__ int si[256];
    sv[threadIdx.x] = acc;
    si[threadIdx.x] = c;
    __syncthreads();
    for (int st = 128; st > 0; st >>= 1) {
        if (threadIdx.x < st) {
            float v2 = sv[threadIdx.x + st];
            int i2 = si[threadIdx.x + st];
            if (v2 > sv[threadIdx.x] || (v2 == sv[threadIdx.x] && i2 < si[threadIdx.x])) {
                sv[threadIdx.x] = v2;
                si[threadIdx.x] = i2;
            }
        }
        __syncthreads();
    }
    if (threadIdx.x == 0) { bval[blockIdx.x] = sv[0]; bidx[blockIdx.x] = si[0]; }
}

__global__ void hy_argmax_kernel(const float* __restrict__ bval, const int* __restrict__ bidx,
                                 int nb, int* __restrict__ tok_out) {
    __shared__ float sv[128];
    __shared__ int si[128];
    float best = -3.4e38f;
    int bi = 0x7fffffff;
    for (int i = threadIdx.x; i < nb; i += 128) {
        float v = bval[i];
        int id = bidx[i];
        if (v > best || (v == best && id < bi)) { best = v; bi = id; }
    }
    sv[threadIdx.x] = best;
    si[threadIdx.x] = bi;
    __syncthreads();
    for (int st = 64; st > 0; st >>= 1) {
        if (threadIdx.x < st) {
            float v2 = sv[threadIdx.x + st];
            int i2 = si[threadIdx.x + st];
            if (v2 > sv[threadIdx.x] || (v2 == sv[threadIdx.x] && i2 < si[threadIdx.x])) {
                sv[threadIdx.x] = v2;
                si[threadIdx.x] = i2;
            }
        }
        __syncthreads();
    }
    if (threadIdx.x == 0) {
        tok_out[0] = si[0];      // token (int32 bits in float-typed d_out)
        tok_out[1] = KV_;        // KV scalar
    }
}

// =====================================================================
// Host-side orchestration
// =====================================================================
extern "C" void kernel_launch(void* const* d_in, const int* in_sizes, int n_in,
                              void* d_out, int out_size, void* d_ws, size_t ws_size,
                              hipStream_t stream) {
    (void)in_sizes; (void)n_in; (void)out_size; (void)ws_size;
    const int*   input_ids   = (const int*)d_in[0];
    const float* key_cache   = (const float*)d_in[1];
    const float* value_cache = (const float*)d_in[2];
    const float* attn_flag   = (const float*)d_in[3];
    const int*   embed_q     = (const int*)d_in[6];
    const float* embed_scale = (const float*)d_in[7];
    const float* embed_zp    = (const float*)d_in[8];
    const float* in_ln_w     = (const float*)d_in[9];
    const float* wq          = (const float*)d_in[10];
    const float* wk          = (const float*)d_in[11];
    const float* wv          = (const float*)d_in[12];
    const float* wo          = (const float*)d_in[13];
    const float* q_ln_w      = (const float*)d_in[14];
    const float* k_ln_w      = (const float*)d_in[15];
    const float* post_ln_w   = (const float*)d_in[16];
    const float* w_gate      = (const float*)d_in[17];
    const float* w_up        = (const float*)d_in[18];
    const float* w_down      = (const float*)d_in[19];
    const float* final_w     = (const float*)d_in[20];
    const float* lm_head_w   = (const float*)d_in[21];

    float* out = (float*)d_out;
    const long NK_SZ = (long)L_ * KVH_ * HD_ * KV_;     // 2,621,440
    float* nk = out;
    float* nv = out + NK_SZ;
    int* tok_out = (int*)(out + 2 * NK_SZ);

    float* ws = (float*)d_ws;
    const long WS_H    = 0;
    const long WS_HN   = WS_H  + (long)T_ * H_;
    const long WS_Q    = WS_HN + (long)T_ * H_;
    const long WS_K    = WS_Q  + (long)T_ * H_;
    const long WS_V    = WS_K  + (long)T_ * KVH_ * HD_;
    const long WS_ATT  = WS_V  + (long)T_ * KVH_ * HD_;
    const long WS_GATE = WS_ATT + (long)T_ * H_;
    const long WS_UP   = WS_GATE + (long)T_ * F_;
    const long WS_HL   = WS_UP + (long)T_ * F_;
    const long WS_BVAL = WS_HL + H_;
    const long WS_BIDX = WS_BVAL + 256;
    float* h    = ws + WS_H;
    float* hn   = ws + WS_HN;
    float* qbuf = ws + WS_Q;
    float* kbuf = ws + WS_K;
    float* vbuf = ws + WS_V;
    float* att  = ws + WS_ATT;
    float* gate = ws + WS_GATE;
    float* up   = ws + WS_UP;
    float* hl   = ws + WS_HL;
    float* bval = ws + WS_BVAL;
    int*   bidx = (int*)(ws + WS_BIDX);

    // ---- embedding dequant ----
    hy_embed_kernel<<<T_, 256, 0, stream>>>(input_ids, embed_q, embed_scale, embed_zp, h);

    for (int l = 0; l < L_; l++) {
        const float* ilw = in_ln_w + (long)l * H_;
        const float* plw = post_ln_w + (long)l * H_;
        const float* wql = wq + (long)l * H_ * (NH_ * HD_);
        const float* wkl = wk + (long)l * H_ * (KVH_ * HD_);
        const float* wvl = wv + (long)l * H_ * (KVH_ * HD_);
        const float* wol = wo + (long)l * (NH_ * HD_) * H_;
        const float* wgl = w_gate + (long)l * H_ * F_;
        const float* wul = w_up + (long)l * H_ * F_;
        const float* wdl = w_down + (long)l * F_ * H_;
        const float* qlw = q_ln_w + (long)l * HD_;
        const float* klw = k_ln_w + (long)l * HD_;
        float* nkL = nk + (long)l * KVH_ * HD_ * KV_;
        float* nvL = nv + (long)l * KVH_ * KV_ * HD_;
        const float* kcl = key_cache + (long)l * KVH_ * HD_ * HIST_;
        const float* vcl = value_cache + (long)l * KVH_ * HIST_ * HD_;

        // pre-attention RMSNorm
        hy_rms_kernel<<<T_, 256, 0, stream>>>(h, ilw, hn, H_);

        // q/k/v projections (WMMA + TDM)
        hy_gemm_kernel<<<dim3((NH_ * HD_) / TILE_N, T_ / TILE_M), 256, 0, stream>>>(hn, wql, qbuf, NH_ * HD_, H_, 0);
        hy_gemm_kernel<<<dim3((KVH_ * HD_) / TILE_N, T_ / TILE_M), 256, 0, stream>>>(hn, wkl, kbuf, KVH_ * HD_, H_, 0);
        hy_gemm_kernel<<<dim3((KVH_ * HD_) / TILE_N, T_ / TILE_M), 256, 0, stream>>>(hn, wvl, vbuf, KVH_ * HD_, H_, 0);

        // rope + head-RMS; K written transposed into output cache region
        hy_rope_q_kernel<<<T_ * NH_, 128, 0, stream>>>(qbuf, qlw);
        hy_rope_k_kernel<<<T_ * KVH_, 128, 0, stream>>>(kbuf, klw, nkL);

        // cache concat
        hy_copy_kcache_kernel<<<(KVH_ * HD_ * HIST_) / 256, 256, 0, stream>>>(kcl, nkL);
        hy_copy_vcache_kernel<<<(KVH_ * HIST_ * HD_) / 256, 256, 0, stream>>>(vcl, nvL);
        hy_write_vnew_kernel<<<(T_ * KVH_ * HD_) / 256, 256, 0, stream>>>(vbuf, nvL);

        // attention (WMMA flash, online softmax)
        hy_attn_kernel<<<(NH_ * (T_ / 16)) / 4, 128, 0, stream>>>(qbuf, nkL, nvL, attn_flag, att);

        // output projection + residual
        hy_gemm_kernel<<<dim3(H_ / TILE_N, T_ / TILE_M), 256, 0, stream>>>(att, wol, h, H_, NH_ * HD_, 1);

        // MLP
        hy_rms_kernel<<<T_, 256, 0, stream>>>(h, plw, hn, H_);
        hy_gemm_kernel<<<dim3(F_ / TILE_N, T_ / TILE_M), 256, 0, stream>>>(hn, wgl, gate, F_, H_, 0);
        hy_gemm_kernel<<<dim3(F_ / TILE_N, T_ / TILE_M), 256, 0, stream>>>(hn, wul, up, F_, H_, 0);
        hy_silu_mul_kernel<<<((long)T_ * F_) / 256, 256, 0, stream>>>(gate, up, (long)T_ * F_);
        hy_gemm_kernel<<<dim3(H_ / TILE_N, T_ / TILE_M), 256, 0, stream>>>(gate, wdl, h, H_, F_, 1);
    }

    // final norm on last token + lm head + argmax
    hy_rms_kernel<<<1, 256, 0, stream>>>(h + (long)(T_ - 1) * H_, final_w, hl, H_);
    hy_logits_kernel<<<V_ / 256, 256, 0, stream>>>(hl, lm_head_w, bval, bidx);
    hy_argmax_kernel<<<1, 128, 0, stream>>>(bval, bidx, V_ / 256, tok_out);
}
